// Head_21827023798616
// MI455X (gfx1250) — matrix operand order repaired and checked by reading, not compiled, not run on hardware
//
#include <hip/hip_runtime.h>
#include <hip/hip_bf16.h>

typedef __attribute__((ext_vector_type(16))) _Float16 v16h;
typedef __attribute__((ext_vector_type(8)))  _Float16 v8h;
typedef __attribute__((ext_vector_type(8)))  float    v8f;

#define E_DIM 512
#define H_DIM 64
#define T_DIM 256
#define B_DIM 256
#define NT    12          // 192 output cols = [k|q|v] as 12 N-tiles of 16
#define CHUNK_ELEMS 6144  // 192 rows x 32 K per staged chunk (12 KB f16)

static __device__ __forceinline__ v8f wmma_f16(v16h a, v16h b, v8f c) {
  // 8 args: (neg_a, A, neg_b, B, c_mod, C, reuse_a, reuse_b)
  return __builtin_amdgcn_wmma_f32_16x16x32_f16(false, a, false, b, (short)0, c,
                                                false, false);
}

// -----------------------------------------------------------------------------
// Kernel 0: convert stacked [Wk;Wq;Wv] fp32 -> f16, pre-chunked as the exact
// LDS image per K-chunk: wst[16 chunks][192 n][32 k]. Makes staging a flat
// 12 KB contiguous async copy.
// -----------------------------------------------------------------------------
__global__ __launch_bounds__(256) void wconv_kernel(
    const float* __restrict__ Wk, const float* __restrict__ Wq,
    const float* __restrict__ Wv, _Float16* __restrict__ wst)
{
  const int c = blockIdx.x * 256 + threadIdx.x;   // 24576 float4 chunks total
  const int n  = c >> 7;                          // 0..191
  const int e4 = (c & 127) << 2;                  // 0,4,...,508
  const float* Wsel = (n < 64) ? Wk : (n < 128) ? Wq : Wv;
  const float4 w = *(const float4*)(Wsel + (size_t)(n & 63) * E_DIM + e4);
  _Float16* d = wst + (size_t)(e4 >> 5) * CHUNK_ELEMS + n * 32 + (e4 & 31);
  d[0] = (_Float16)w.x; d[1] = (_Float16)w.y;
  d[2] = (_Float16)w.z; d[3] = (_Float16)w.w;
}

// -----------------------------------------------------------------------------
// Kernel 1: fused K/Q/V projection. Block = 128 rows of flattened [B*T, E],
// 8 waves x (16 rows x 192 cols). Weight chunks double-buffered in LDS via
// async global->LDS copies (ASYNCcnt) overlapped with the WMMA K-loop.
// V is written transposed [B, 64, T] for the attention kernel's B-fragments.
// -----------------------------------------------------------------------------
__global__ __launch_bounds__(256) void proj_kernel(
    const float* __restrict__ x, const _Float16* __restrict__ wst,
    _Float16* __restrict__ kh, _Float16* __restrict__ qh,
    _Float16* __restrict__ vT)
{
  __shared__ __attribute__((aligned(32))) _Float16 sW[2][CHUNK_ELEMS]; // 24 KB

  const int tid  = threadIdx.x;
  const int wave = tid >> 5;
  const int lane = tid & 31;
  const int l16  = lane & 15;
  const int hi   = lane >> 4;   // which 16-lane half
  const int c0   = hi * 16;     // A-frag K sub-offset: contiguous k = 16*hi + s

  const int rowA = blockIdx.x * 128 + wave * 16 + l16;  // this lane's A row

  // Issue this thread's share (3 x b128) of one 12 KB chunk -> LDS buffer.
  // Generic shared pointer low 32 bits == LDS byte offset (LDS aperture form).
  auto issue = [&](int chunk, int buf) {
    const uint64_t g = (uint64_t)(uintptr_t)(wst + (size_t)chunk * CHUNK_ELEMS)
                     + (uint32_t)(tid * 16);
    const uint32_t l = (uint32_t)(uintptr_t)(&sW[buf][0]) + (uint32_t)(tid * 16);
    #pragma unroll
    for (int j = 0; j < 3; ++j) {
      asm volatile("global_load_async_to_lds_b128 %0, %1, off"
                   :: "v"(l + (uint32_t)(j * 4096)), "v"(g + (uint64_t)(j * 4096))
                   : "memory");
    }
  };

  v8f acc[NT];
  #pragma unroll
  for (int n = 0; n < NT; ++n)
    #pragma unroll
    for (int e = 0; e < 8; ++e) acc[n][e] = 0.0f;

  issue(0, 0);                                  // prefetch chunk 0

  for (int i = 0; i < 16; ++i) {                // 16 K-chunks of 32
    const int kk = i * 32;
    if (i + 1 < 16) {
      issue(i + 1, (i + 1) & 1);                // prefetch next chunk
      asm volatile("s_wait_asynccnt 0x3" ::: "memory");  // current chunk landed
    } else {
      asm volatile("s_wait_asynccnt 0x0" ::: "memory");
    }
    __syncthreads();                            // all waves' copies visible

    // A fragment: 16 contiguous fp32 at K=kk+c0, cvt to f16.
    v16h a;
    {
      const float* p = x + (size_t)rowA * E_DIM + kk + c0;
      const float4 f0 = *(const float4*)(p);
      const float4 f1 = *(const float4*)(p + 4);
      const float4 f2 = *(const float4*)(p + 8);
      const float4 f3 = *(const float4*)(p + 12);
      a[0]=(_Float16)f0.x; a[1]=(_Float16)f0.y; a[2]=(_Float16)f0.z; a[3]=(_Float16)f0.w;
      a[4]=(_Float16)f1.x; a[5]=(_Float16)f1.y; a[6]=(_Float16)f1.z; a[7]=(_Float16)f1.w;
      a[8]=(_Float16)f2.x; a[9]=(_Float16)f2.y; a[10]=(_Float16)f2.z; a[11]=(_Float16)f2.w;
      a[12]=(_Float16)f3.x; a[13]=(_Float16)f3.y; a[14]=(_Float16)f3.z; a[15]=(_Float16)f3.w;
    }

    // One A fragment feeds all 12 N-tiles (reuse => high arithmetic intensity).
    const _Float16* sWc = &sW[i & 1][0];
    #pragma unroll
    for (int n = 0; n < NT; ++n) {
      const v16h b = *(const v16h*)&sWc[(n * 16 + l16) * 32 + hi * 16];
      acc[n] = wmma_f16(a, b, acc[n]);
    }

    __syncthreads();   // reads done before this buffer is async-overwritten
  }

  // Epilogue. C layout: lane holds col N=l16, rows M = r + 8*hi.
  const int rowC0 = blockIdx.x * 128 + wave * 16 + 8 * hi;
  #pragma unroll
  for (int n = 0; n < 8; ++n) {                 // k (n 0..3), q (n 4..7)
    _Float16* dst = (n < 4) ? kh : qh;
    const int col = (n & 3) * 16 + l16;
    #pragma unroll
    for (int r = 0; r < 8; ++r)
      dst[(size_t)(rowC0 + r) * H_DIM + col] = (_Float16)acc[n][r];
  }
  #pragma unroll
  for (int n = 8; n < NT; ++n) {                // v, transposed [B,64,T]
    const int hcol = (n - 8) * 16 + l16;
    #pragma unroll
    for (int r = 0; r < 8; ++r) {
      const int grow = rowC0 + r;
      const int bb = grow >> 8;
      const int tt = grow & 255;
      vT[(size_t)(bb * H_DIM + hcol) * T_DIM + tt] = (_Float16)acc[n][r];
    }
  }
}

// -----------------------------------------------------------------------------
// Kernel 2: flash-style causal attention. Wave = one 16-row Q tile; key loop in
// blocks of 32 with online softmax; P re-laid-out C->A via per-wave LDS slab.
// -----------------------------------------------------------------------------
__global__ __launch_bounds__(256) void attn_kernel(
    const _Float16* __restrict__ qh, const _Float16* __restrict__ kh,
    const _Float16* __restrict__ vT, float* __restrict__ out)
{
  __shared__ __attribute__((aligned(32))) _Float16 sP[8 * 16 * 32]; // 8 KB

  const int tid  = threadIdx.x;
  const int wave = tid >> 5;
  const int lane = tid & 31;
  const int l16  = lane & 15;
  const int hi   = lane >> 4;

  const int b  = blockIdx.x >> 1;
  const int q0 = (blockIdx.x & 1) * 128 + wave * 16;

  const _Float16* qb = qh + (size_t)b * T_DIM * H_DIM;
  const _Float16* kb = kh + (size_t)b * T_DIM * H_DIM;
  const _Float16* vb = vT + (size_t)b * H_DIM * T_DIM;

  // Q A-fragments over head dim (two 16x32 chunks).
  v16h qf[2];
  #pragma unroll
  for (int j = 0; j < 2; ++j) {
    const _Float16* p = qb + (size_t)(q0 + l16) * H_DIM + j * 32 + hi * 16;
    const v8h lo = *(const v8h*)(p);
    const v8h hh = *(const v8h*)(p + 8);
    #pragma unroll
    for (int e = 0; e < 8; ++e) { qf[j][e] = lo[e]; qf[j][e + 8] = hh[e]; }
  }

  v8f o[4];
  float m_i[8], l_i[8];
  #pragma unroll
  for (int t = 0; t < 4; ++t)
    #pragma unroll
    for (int e = 0; e < 8; ++e) o[t][e] = 0.0f;
  #pragma unroll
  for (int r = 0; r < 8; ++r) { m_i[r] = -1e30f; l_i[r] = 0.0f; }

  _Float16* sPw = &sP[wave * 512];
  const float scale = 0.125f;   // 64^-0.5

  for (int s0 = 0; s0 <= q0 + 15; s0 += 32) {   // wave-uniform causal bound
    // S = Q * K^T : two 16x16 key tiles, K-dim 64 = two WMMAs each.
    v8f st[2];
    #pragma unroll
    for (int n2 = 0; n2 < 2; ++n2) {
      #pragma unroll
      for (int e = 0; e < 8; ++e) st[n2][e] = 0.0f;
      #pragma unroll
      for (int j = 0; j < 2; ++j) {
        const v16h bk = *(const v16h*)(kb +
            (size_t)(s0 + n2 * 16 + l16) * H_DIM + j * 32 + hi * 16);
        st[n2] = wmma_f16(qf[j], bk, st[n2]);
      }
    }

    float sv0[8], sv1[8], rmax[8];
    const bool need_mask = (s0 + 31 > q0);
    #pragma unroll
    for (int r = 0; r < 8; ++r) {
      const int qrow = q0 + r + 8 * hi;
      float a0 = st[0][r] * scale;
      float a1 = st[1][r] * scale;
      if (need_mask) {
        if (s0 + l16 > qrow)      a0 = -1e30f;
        if (s0 + 16 + l16 > qrow) a1 = -1e30f;
      }
      sv0[r] = a0; sv1[r] = a1;
      rmax[r] = fmaxf(a0, a1);
    }
    // Row max across the 16 lanes holding this row (xor stays within halves).
    #pragma unroll
    for (int msk = 1; msk <= 8; msk <<= 1)
      #pragma unroll
      for (int r = 0; r < 8; ++r)
        rmax[r] = fmaxf(rmax[r], __shfl_xor(rmax[r], msk, 32));

    #pragma unroll
    for (int r = 0; r < 8; ++r) {
      const float mn = fmaxf(m_i[r], rmax[r]);
      const float al = __expf(m_i[r] - mn);
      m_i[r] = mn;
      const float p0 = __expf(sv0[r] - mn);
      const float p1 = __expf(sv1[r] - mn);
      l_i[r] = l_i[r] * al + p0 + p1;       // per-lane partial; reduced at end
      #pragma unroll
      for (int t = 0; t < 4; ++t) o[t][r] *= al;
      // Scatter P (C layout) to LDS in [M][K] order.
      sPw[(r + 8 * hi) * 32 + l16]      = (_Float16)p0;
      sPw[(r + 8 * hi) * 32 + 16 + l16] = (_Float16)p1;
    }

    // Same-wave LDS RAW: DS ops are in-order; fence the counter explicitly.
    asm volatile("s_wait_dscnt 0x0" ::: "memory");

    // Gather P back in A-fragment layout.
    v16h pa;
    {
      const _Float16* p = &sPw[l16 * 32 + hi * 16];
      const v8h lo = *(const v8h*)(p);
      const v8h hh = *(const v8h*)(p + 8);
      #pragma unroll
      for (int e = 0; e < 8; ++e) { pa[e] = lo[e]; pa[e + 8] = hh[e]; }
    }

    // O += P * V : vT gives contiguous 32-byte B-fragments.
    #pragma unroll
    for (int h4 = 0; h4 < 4; ++h4) {
      const v16h bv = *(const v16h*)(vb +
          (size_t)(h4 * 16 + l16) * T_DIM + s0 + hi * 16);
      o[h4] = wmma_f16(pa, bv, o[h4]);
    }
  }

  // Reduce l across the 16 lanes of each row, then write out / l.
  #pragma unroll
  for (int msk = 1; msk <= 8; msk <<= 1)
    #pragma unroll
    for (int r = 0; r < 8; ++r)
      l_i[r] += __shfl_xor(l_i[r], msk, 32);

  float* ob = out + (size_t)b * T_DIM * H_DIM;
  #pragma unroll
  for (int r = 0; r < 8; ++r) {
    const float inv = 1.0f / l_i[r];
    const int row = q0 + r + 8 * hi;
    #pragma unroll
    for (int h4 = 0; h4 < 4; ++h4)
      ob[(size_t)row * H_DIM + h4 * 16 + l16] = o[h4][r] * inv;
  }
}

// -----------------------------------------------------------------------------
extern "C" void kernel_launch(void* const* d_in, const int* in_sizes, int n_in,
                              void* d_out, int out_size, void* d_ws, size_t ws_size,
                              hipStream_t stream) {
  (void)in_sizes; (void)n_in; (void)out_size; (void)ws_size;
  const float* x  = (const float*)d_in[0];
  const float* Wk = (const float*)d_in[1];
  const float* Wq = (const float*)d_in[2];
  const float* Wv = (const float*)d_in[3];
  float* out = (float*)d_out;

  const size_t elems = (size_t)B_DIM * T_DIM * H_DIM;   // 4M elems per tensor
  _Float16* kh  = (_Float16*)d_ws;                      // [B*T, 64] f16
  _Float16* qh  = kh + elems;                           // [B*T, 64] f16
  _Float16* vT  = qh + elems;                           // [B, 64, T] f16
  _Float16* wst = vT + elems;                           // [16][192][32] f16

  wconv_kernel<<<96, 256, 0, stream>>>(Wk, Wq, Wv, wst);
  proj_kernel<<<512, 256, 0, stream>>>(x, wst, kh, qh, vT);
  attn_kernel<<<512, 256, 0, stream>>>(qh, kh, vT, out);
}
